// WindowedMHSA_10239202033899
// MI455X (gfx1250) — compile-verified, matrix operand order checked by reading, not run on hardware
//
#include <hip/hip_runtime.h>
#include <hip/hip_bf16.h>
#include <stdint.h>

// ---------------- problem constants ----------------
#define Cdim   512
#define Tdim   16
#define Hdim   56
#define Wdim   56
#define WTm    2
#define WHm    7
#define WWm    7
#define Ntok   98          // WT*WH*WW
#define Npad   112         // 7*16, window-aligned token padding
#define NWIN   512         // 8*8*8 windows
#define NHd    8           // heads
#define HD     64          // head dim
#define QKVN   1536        // 3*C
#define Mtot   (NWIN*Npad) // 57344 padded token rows

typedef __attribute__((ext_vector_type(16))) __bf16 v16bf;
typedef __attribute__((ext_vector_type(8)))  float  v8f;

union V16 { uint4 u[2]; v16bf v; };

__device__ __forceinline__ unsigned short f2bf(float f) {
  unsigned u = __float_as_uint(f);
  u += 0x7FFFu + ((u >> 16) & 1u);   // round-to-nearest-even
  return (unsigned short)(u >> 16);
}

__device__ __forceinline__ v8f wmma_bf16(const V16& a, const V16& b, v8f c) {
  return __builtin_amdgcn_wmma_f32_16x16x32_bf16(false, a.v, false, b.v,
                                                 (short)0, c, false, false);
}

// ---------------- K0: window partition + fp32->bf16 ----------------
// xw layout: (win, ntok[0..111], c) bf16, pad rows zeroed
__global__ void pack_x_kernel(const float* __restrict__ x,
                              unsigned short* __restrict__ xw) {
  long idx = (long)blockIdx.x * blockDim.x + threadIdx.x;
  if (idx >= (long)Mtot * Cdim) return;
  int c    = (int)(idx & (Cdim - 1));
  int gm   = (int)(idx >> 9);
  int ntok = gm % Npad;
  int win  = gm / Npad;
  unsigned short val = 0;
  if (ntok < Ntok) {
    int wt = ntok / 49, rem = ntok % 49, wh = rem / 7, ww = rem % 7;
    int tx = win & 7, ty = (win >> 3) & 7, tz = win >> 6;
    int t = tz * WTm + wt, h = ty * WHm + wh, w = tx * WWm + ww;
    float f = x[(long)c * (Tdim * Hdim * Wdim) + (long)t * (Hdim * Wdim) + h * Wdim + w];
    val = f2bf(f);
  }
  xw[idx] = val;
}

// ---------------- K1: weight transpose fp32(K,N) -> bf16(N,K) ----------------
__global__ void pack_wT_kernel(const float* __restrict__ w,
                               unsigned short* __restrict__ wT, int K, int N) {
  long idx = (long)blockIdx.x * blockDim.x + threadIdx.x;
  if (idx >= (long)K * N) return;
  int k = (int)(idx / N), n = (int)(idx % N);
  wT[(long)n * K + k] = f2bf(w[idx]);
}

// ---------------- K2: QKV GEMM (bf16 WMMA) + bias + scatter to q/k/vT ----------------
// one wave -> 32x64 output (2 M-tiles x 4 N-tiles); K=512 in 16 steps of 32.
// B fragments are reused across the two M-tiles.
__global__ void __launch_bounds__(256)
qkv_gemm_kernel(const unsigned short* __restrict__ xw,
                const unsigned short* __restrict__ wqkvT,
                const float* __restrict__ b_qkv,
                unsigned short* __restrict__ qb,
                unsigned short* __restrict__ kb,
                unsigned short* __restrict__ vT) {
  int wid = blockIdx.x * 8 + (threadIdx.x >> 5);
  int mt = wid / 24;          // 1792 double row tiles (32 rows each)
  int ng = wid % 24;          // 24 groups of 64 cols
  int lane = threadIdx.x & 31;
  int r = lane & 15, hs = lane >> 4;

  v8f acc[2][4] = {};
  const unsigned short* Arow0 = xw + (long)(mt * 32 + r) * Cdim;
  const unsigned short* Arow1 = Arow0 + 16 * Cdim;
#pragma unroll 4
  for (int kk = 0; kk < 16; ++kk) {
    __builtin_prefetch(Arow0 + (kk + 2) * 32, 0, 3);
    __builtin_prefetch(Arow1 + (kk + 2) * 32, 0, 3);
    V16 a0, a1;
    a0.u[0] = *(const uint4*)(Arow0 + kk * 32 + hs * 8);
    a0.u[1] = *(const uint4*)(Arow0 + kk * 32 + 16 + hs * 8);
    a1.u[0] = *(const uint4*)(Arow1 + kk * 32 + hs * 8);
    a1.u[1] = *(const uint4*)(Arow1 + kk * 32 + 16 + hs * 8);
#pragma unroll
    for (int j = 0; j < 4; ++j) {
      const unsigned short* Bcol = wqkvT + (long)(ng * 64 + j * 16 + r) * Cdim;
      V16 b;
      b.u[0] = *(const uint4*)(Bcol + kk * 32 + hs * 8);
      b.u[1] = *(const uint4*)(Bcol + kk * 32 + 16 + hs * 8);
      acc[0][j] = wmma_bf16(a0, b, acc[0][j]);
      acc[1][j] = wmma_bf16(a1, b, acc[1][j]);
    }
  }
#pragma unroll
  for (int t = 0; t < 2; ++t) {
#pragma unroll
    for (int j = 0; j < 4; ++j) {
      int gc = ng * 64 + j * 16 + r;
      int which = gc >> 9;              // 0=q 1=k 2=v
      int hrem = gc & 511;
      int head = hrem >> 6, d = hrem & 63;
      float bias = b_qkv[gc];
#pragma unroll
      for (int i = 0; i < 8; ++i) {
        int gm = mt * 32 + t * 16 + i + 8 * hs;
        int win = gm / Npad, ntok = gm % Npad;
        float val = acc[t][j][i] + bias;
        long wh = (long)(win * NHd + head);
        if (which == 0) {
          val *= 0.125f;  // 1/sqrt(64) folded into q
          qb[(wh * Npad + ntok) * HD + d] = f2bf(val);
        } else if (which == 1) {
          kb[(wh * Npad + ntok) * HD + d] = f2bf(val);
        } else {
          vT[(wh * HD + d) * Npad + ntok] = f2bf(val);  // transposed for P@v B-operand
        }
      }
    }
  }
}

// ---------------- K3: attention per (window, head) ----------------
// block = 224 threads (7 waves), wave w owns rows [16w, 16w+16)
__global__ void __launch_bounds__(224)
attn_kernel(const unsigned short* __restrict__ qb,
            const unsigned short* __restrict__ kb,
            const unsigned short* __restrict__ vT,
            unsigned short* __restrict__ ao) {
  __shared__ float          S[7][16][Npad];  // logits
  __shared__ unsigned short P[7][16][128];   // probs bf16, K padded to 128

  int bid = blockIdx.x;               // win*8 + head
  int win = bid >> 3, head = bid & 7;
  int wv = threadIdx.x >> 5;
  int lane = threadIdx.x & 31;
  int r = lane & 15, hs = lane >> 4;

  const unsigned short* qbase = qb + (long)bid * Npad * HD;
  const unsigned short* kbase = kb + (long)bid * Npad * HD;
  const unsigned short* vbase = vT + (long)bid * HD * Npad;

  // ---- S = q @ k^T (pre-scaled q) ----
  const unsigned short* Arow = qbase + (wv * 16 + r) * HD;
#pragma unroll
  for (int ct = 0; ct < 7; ++ct) {
    v8f s = {};
#pragma unroll
    for (int kk = 0; kk < 2; ++kk) {
      V16 a, b;
      a.u[0] = *(const uint4*)(Arow + kk * 32 + hs * 8);
      a.u[1] = *(const uint4*)(Arow + kk * 32 + 16 + hs * 8);
      const unsigned short* Bcol = kbase + (ct * 16 + r) * HD;
      b.u[0] = *(const uint4*)(Bcol + kk * 32 + hs * 8);
      b.u[1] = *(const uint4*)(Bcol + kk * 32 + 16 + hs * 8);
      s = wmma_bf16(a, b, s);
    }
#pragma unroll
    for (int i = 0; i < 8; ++i) S[wv][i + 8 * hs][ct * 16 + r] = s[i];
  }
  __syncthreads();

  // ---- masked softmax over key dim (cols >= 98 masked), full wave ----
  // lane half hs handles columns [hs*56, hs*56+56) of row (lane&15);
  // cross-half combine via shfl_xor(16).
  {
    int c0 = hs * 56;
    int cend = hs ? Ntok : 56;
    float mx = -1e30f;
    for (int c2 = c0; c2 < cend; ++c2) mx = fmaxf(mx, S[wv][r][c2]);
    mx = fmaxf(mx, __shfl_xor(mx, 16));
    float sum = 0.f;
    for (int c2 = c0; c2 < cend; ++c2) {
      float e = __expf(S[wv][r][c2] - mx);
      sum += e;
      S[wv][r][c2] = e;
    }
    sum += __shfl_xor(sum, 16);
    float inv = 1.0f / sum;
    __syncthreads();
    // lane half hs fills P columns [hs*64, hs*64+64)
    for (int c2 = hs * 64; c2 < hs * 64 + 64; ++c2)
      P[wv][r][c2] = (c2 < Ntok) ? f2bf(S[wv][r][c2] * inv) : (unsigned short)0;
  }
  __syncthreads();

  // ---- O = P @ v, K = 128 (padded) in 4 steps ----
  v8f o[4] = {};
#pragma unroll
  for (int kk = 0; kk < 4; ++kk) {
    V16 a;
    a.u[0] = *(const uint4*)(&P[wv][r][kk * 32 + hs * 8]);
    a.u[1] = *(const uint4*)(&P[wv][r][kk * 32 + 16 + hs * 8]);
    int t0 = kk * 32 + hs * 8;
    int t1 = kk * 32 + 16 + hs * 8;
#pragma unroll
    for (int j = 0; j < 4; ++j) {
      const unsigned short* Brow = vbase + (j * 16 + r) * Npad;  // tokens contiguous
      V16 b;
      b.u[0] = (t0 + 7 < Npad) ? *(const uint4*)(Brow + t0) : make_uint4(0, 0, 0, 0);
      b.u[1] = (t1 + 7 < Npad) ? *(const uint4*)(Brow + t1) : make_uint4(0, 0, 0, 0);
      o[j] = wmma_bf16(a, b, o[j]);
    }
  }
  // write attention output (win, ntok, head*64+d) bf16
#pragma unroll
  for (int j = 0; j < 4; ++j) {
#pragma unroll
    for (int i = 0; i < 8; ++i) {
      int ntok = wv * 16 + i + 8 * hs;
      long gm = (long)win * Npad + ntok;
      ao[gm * Cdim + head * 64 + j * 16 + r] = f2bf(o[j][i]);
    }
  }
}

// ---------------- K4: proj GEMM + bias + window reverse scatter ----------------
// one wave -> 32x64 output (2 M-tiles x 4 N-tiles), B reused across M-tiles
__global__ void __launch_bounds__(256)
proj_gemm_kernel(const unsigned short* __restrict__ ao,
                 const unsigned short* __restrict__ wprojT,
                 const float* __restrict__ b_proj,
                 float* __restrict__ out) {
  int wid = blockIdx.x * 8 + (threadIdx.x >> 5);
  int mt = wid / 8;           // 1792 double row tiles
  int ng = wid % 8;           // 8 groups of 64 cols
  int lane = threadIdx.x & 31;
  int r = lane & 15, hs = lane >> 4;

  v8f acc[2][4] = {};
  const unsigned short* Arow0 = ao + (long)(mt * 32 + r) * Cdim;
  const unsigned short* Arow1 = Arow0 + 16 * Cdim;
#pragma unroll 4
  for (int kk = 0; kk < 16; ++kk) {
    __builtin_prefetch(Arow0 + (kk + 2) * 32, 0, 3);
    __builtin_prefetch(Arow1 + (kk + 2) * 32, 0, 3);
    V16 a0, a1;
    a0.u[0] = *(const uint4*)(Arow0 + kk * 32 + hs * 8);
    a0.u[1] = *(const uint4*)(Arow0 + kk * 32 + 16 + hs * 8);
    a1.u[0] = *(const uint4*)(Arow1 + kk * 32 + hs * 8);
    a1.u[1] = *(const uint4*)(Arow1 + kk * 32 + 16 + hs * 8);
#pragma unroll
    for (int j = 0; j < 4; ++j) {
      const unsigned short* Bcol = wprojT + (long)(ng * 64 + j * 16 + r) * Cdim;
      V16 b;
      b.u[0] = *(const uint4*)(Bcol + kk * 32 + hs * 8);
      b.u[1] = *(const uint4*)(Bcol + kk * 32 + 16 + hs * 8);
      acc[0][j] = wmma_bf16(a0, b, acc[0][j]);
      acc[1][j] = wmma_bf16(a1, b, acc[1][j]);
    }
  }
#pragma unroll
  for (int t = 0; t < 2; ++t) {
#pragma unroll
    for (int j = 0; j < 4; ++j) {
      int gc = ng * 64 + j * 16 + r;       // output channel
      float bias = b_proj[gc];
#pragma unroll
      for (int i = 0; i < 8; ++i) {
        int gm = mt * 32 + t * 16 + i + 8 * hs;
        int win = gm / Npad, ntok = gm % Npad;
        if (ntok < Ntok) {
          int wt = ntok / 49, rem = ntok % 49, wh = rem / 7, ww = rem % 7;
          int tx = win & 7, ty = (win >> 3) & 7, tz = win >> 6;
          int t2 = tz * WTm + wt, h = ty * WHm + wh, w = tx * WWm + ww;
          out[(long)gc * (Tdim * Hdim * Wdim) + (long)t2 * (Hdim * Wdim) + h * Wdim + w] =
              acc[t][j][i] + bias;
        }
      }
    }
  }
}

// ---------------- host launch ----------------
extern "C" void kernel_launch(void* const* d_in, const int* in_sizes, int n_in,
                              void* d_out, int out_size, void* d_ws, size_t ws_size,
                              hipStream_t stream) {
  const float* x      = (const float*)d_in[0];
  const float* w_qkv  = (const float*)d_in[1];
  const float* b_qkv  = (const float*)d_in[2];
  const float* w_proj = (const float*)d_in[3];
  const float* b_proj = (const float*)d_in[4];
  float* out = (float*)d_out;

  char* base = (char*)d_ws;
  size_t off = 0;
  const size_t big = (size_t)Mtot * Cdim * 2;          // 58,720,256 B
  unsigned short* xw     = (unsigned short*)(base + off); off += big;
  unsigned short* wqkvT  = (unsigned short*)(base + off); off += (size_t)QKVN * Cdim * 2;
  unsigned short* wprojT = (unsigned short*)(base + off); off += (size_t)Cdim * Cdim * 2;
  unsigned short* qb     = (unsigned short*)(base + off); off += big;
  unsigned short* kb     = (unsigned short*)(base + off); off += big;
  unsigned short* vT     = (unsigned short*)(base + off); off += big;
  unsigned short* ao     = (unsigned short*)(base + off); off += big;
  (void)ws_size; (void)in_sizes; (void)n_in; (void)out_size;

  // K0: window partition + bf16 convert
  {
    long total = (long)Mtot * Cdim;
    int blocks = (int)((total + 255) / 256);
    pack_x_kernel<<<blocks, 256, 0, stream>>>(x, xw);
  }
  // K1: weight transposes
  pack_wT_kernel<<<(Cdim * QKVN + 255) / 256, 256, 0, stream>>>(w_qkv, wqkvT, Cdim, QKVN);
  pack_wT_kernel<<<(Cdim * Cdim + 255) / 256, 256, 0, stream>>>(w_proj, wprojT, Cdim, Cdim);
  // K2: QKV GEMM  (1792 double-tiles * 24 col-groups = 43008 waves / 8 per block)
  qkv_gemm_kernel<<<5376, 256, 0, stream>>>(xw, wqkvT, b_qkv, qb, kb, vT);
  // K3: attention, one block per (window, head)
  attn_kernel<<<NWIN * NHd, 224, 0, stream>>>(qb, kb, vT, ao);
  // K4: proj GEMM + window reverse (1792 double-tiles * 8 col-groups / 8 waves per block)
  proj_gemm_kernel<<<1792, 256, 0, stream>>>(ao, wprojT, b_proj, out);
}
